// EnoughViTEncoder_63282048139393
// MI455X (gfx1250) — matrix operand order repaired and checked by reading, not compiled, not run on hardware
//
#include <hip/hip_runtime.h>
#include <hip/hip_bf16.h>

typedef __attribute__((ext_vector_type(16))) __bf16 v16bf;
typedef __attribute__((ext_vector_type(8)))  __bf16 v8bf;
typedef __attribute__((ext_vector_type(8)))  float  v8f;
typedef int v4i __attribute__((vector_size(16)));

static constexpr int Dm   = 1024;   // model dim
static constexpr int Sm   = 4096;   // sequence
static constexpr int Bm_  = 4;      // batch
static constexpr int ROWS = Sm * Bm_;

// ---- CDNA5 async global->LDS path selection ---------------------------------
#if defined(__gfx1250__) && __has_builtin(__builtin_amdgcn_global_load_async_to_lds_b128)
#define ASYNC_MODE 2   // clang builtin (probe-confirmed to exist; takes v4i* args)
#elif defined(__gfx1250__)
#define ASYNC_MODE 1   // inline asm
#else
#define ASYNC_MODE 0   // host pass / fallback: synchronous copy
#endif

__device__ __forceinline__ void cp_async16(const __bf16* g, __bf16* l) {
#if ASYNC_MODE == 2
  typedef __attribute__((address_space(1))) v4i* gp_t;
  typedef __attribute__((address_space(3))) v4i* lp_t;
  __builtin_amdgcn_global_load_async_to_lds_b128((gp_t)(g), (lp_t)(l), 0, 0);
#elif ASYNC_MODE == 1
  unsigned lds_addr = (unsigned)(uintptr_t)l;
  asm volatile("global_load_async_to_lds_b128 %0, %1, off"
               :: "v"(lds_addr), "v"(g) : "memory");
#else
  *(uint4*)l = *(const uint4*)g;
#endif
}

__device__ __forceinline__ void wait_async_lds() {
#if ASYNC_MODE == 2
#if __has_builtin(__builtin_amdgcn_s_wait_asynccnt)
  __builtin_amdgcn_s_wait_asynccnt(0);
#else
  asm volatile("s_wait_asynccnt 0x0" ::: "memory");
#endif
#elif ASYNC_MODE == 1
  asm volatile("s_wait_asynccnt 0x0" ::: "memory");
#endif
}

// ---------------- GEMM: C[m,n] = sum_k A[m,k] * B[n,k]  (bf16 in, f32 acc) ----
// Block tile 256x128x32, 256 threads = 8 wave32s in 4x2; wave tile 64x64.
// Double-buffered LDS, async global->LDS pipelined against WMMA.
static constexpr int BM = 256, BN = 128, BK = 32, BKP = BK + 8;

enum { EPI_BF16_T = 0, EPI_GELU_BF16 = 1, EPI_F32_RESID = 2 };

__device__ __forceinline__ v16bf load_fragA(const __bf16* rowbase, int half) {
  // A 16x32 bf16 layout: elems[0..7]=K half*8..+7, elems[8..15]=K 16+half*8..+7
  v8bf lo = *(const v8bf*)(rowbase + half * 8);
  v8bf hi = *(const v8bf*)(rowbase + 16 + half * 8);
  v16bf r;
#pragma unroll
  for (int j = 0; j < 8; ++j) { r[j] = lo[j]; r[j + 8] = hi[j]; }
  return r;
}

__device__ __forceinline__ v16bf load_fragB(const __bf16* colbase, int half) {
  // B 32x16 bf16 layout: elems[0..7]=K half*8..+7, elems[8..15]=K 16+half*8..+7
  v8bf c0 = *(const v8bf*)(colbase + half * 8);
  v8bf c1 = *(const v8bf*)(colbase + 16 + half * 8);
  v16bf r;
#pragma unroll
  for (int j = 0; j < 8; ++j) { r[j] = c0[j]; r[j + 8] = c1[j]; }
  return r;
}

template <int MODE>
__global__ __launch_bounds__(256) void gemm_bf16_wmma(
    const __bf16* __restrict__ A, int lda,
    const __bf16* __restrict__ B, int ldb, int K,
    float* __restrict__ outF, __bf16* __restrict__ outB, int ldc,
    const float* __restrict__ bias, const float* __restrict__ resid, int ldres,
    const float* __restrict__ colmean, float alpha) {
  __shared__ __bf16 As[2][BM][BKP];
  __shared__ __bf16 Bs[2][BN][BKP];

  const int tid   = threadIdx.x;
  const int lane  = tid & 31;
  const int half  = lane >> 4;
  const int cn    = lane & 15;
  const int wave  = tid >> 5;          // 0..7
  const int waveM = wave >> 1;         // 0..3  (64-row slab)
  const int waveN = wave & 1;          // 0..1  (64-col slab)
  const long bm0  = (long)blockIdx.y * BM;
  const long bn0  = (long)blockIdx.x * BN;

  v8f acc[4][4];
#pragma unroll
  for (int sm = 0; sm < 4; ++sm)
#pragma unroll
    for (int sn = 0; sn < 4; ++sn)
#pragma unroll
      for (int v = 0; v < 8; ++v) acc[sm][sn][v] = 0.0f;

  auto issue_tile = [&](int k0, int buf) {
    // A tile: 256 rows x 32 cols = 4 x b128 per thread
#pragma unroll
    for (int i = 0; i < 4; ++i) {
      int idx = tid + i * 256;
      int row = idx >> 2;
      int c8  = (idx & 3) << 3;
      cp_async16(A + (bm0 + row) * (long)lda + k0 + c8, &As[buf][row][c8]);
    }
    // B tile: 128 rows x 32 cols = 2 x b128 per thread
#pragma unroll
    for (int i = 0; i < 2; ++i) {
      int idx = tid + i * 256;
      int row = idx >> 2;
      int c8  = (idx & 3) << 3;
      cp_async16(B + (bn0 + row) * (long)ldb + k0 + c8, &Bs[buf][row][c8]);
    }
  };

  const int T = K / BK;
  issue_tile(0, 0);

  for (int t = 0; t < T; ++t) {
    wait_async_lds();       // my slab-t async copies have landed in LDS
    __syncthreads();        // everyone's landed; prev buffer free for reuse
    if (t + 1 < T) issue_tile((t + 1) * BK, (t + 1) & 1);  // overlap with WMMA

    const int buf = t & 1;
    v16bf af[4], bfv[4];
#pragma unroll
    for (int sm = 0; sm < 4; ++sm)
      af[sm] = load_fragA(&As[buf][waveM * 64 + sm * 16 + cn][0], half);
#pragma unroll
    for (int sn = 0; sn < 4; ++sn)
      bfv[sn] = load_fragB(&Bs[buf][waveN * 64 + sn * 16 + cn][0], half);

#pragma unroll
    for (int sm = 0; sm < 4; ++sm)
#pragma unroll
      for (int sn = 0; sn < 4; ++sn)
        acc[sm][sn] = __builtin_amdgcn_wmma_f32_16x16x32_bf16(
            false, af[sm], false, bfv[sn], (short)0, acc[sm][sn], false, false);
  }

  // epilogue: C lane mapping  M = v + 8*half, N = cn
#pragma unroll
  for (int sm = 0; sm < 4; ++sm) {
#pragma unroll
    for (int sn = 0; sn < 4; ++sn) {
      long gmB = bm0 + waveM * 64 + sm * 16 + half * 8;
      long gn  = bn0 + waveN * 64 + sn * 16 + cn;
#pragma unroll
      for (int v = 0; v < 8; ++v) {
        long gm = gmB + v;
        float val = acc[sm][sn][v] * alpha;
        if (MODE == EPI_BF16_T) {
          outB[gn * (long)ldc + gm] = (__bf16)val;
        } else if (MODE == EPI_GELU_BF16) {
          float t = val + bias[gn];
          float g = 0.5f * t * (1.0f + erff(t * 0.70710678118654752f));
          outB[gm * (long)ldc + gn] = (__bf16)g;
        } else {
          float t = val;
          if (colmean) t += colmean[gn];
          if (bias)    t += bias[gn];
          t += resid[gm * (long)ldres + gn];
          outF[gm * (long)ldc + gn] = t;
        }
      }
    }
  }
}

// ---------------- LayerNorm: one block per row of D=1024 ----------------------
__global__ __launch_bounds__(256) void ln_kernel(
    const float* __restrict__ x, const float* __restrict__ g,
    const float* __restrict__ b, float* __restrict__ outF,
    __bf16* __restrict__ outB) {
  __shared__ float s1[256], s2[256];
  const long row = blockIdx.x;
  const int  tid = threadIdx.x;
  const float* xr = x + row * (long)Dm;

  float xv[4], s = 0.f, sq = 0.f;
#pragma unroll
  for (int i = 0; i < 4; ++i) {
    xv[i] = xr[tid + i * 256];
    s += xv[i]; sq += xv[i] * xv[i];
  }
  s1[tid] = s; s2[tid] = sq;
  __syncthreads();
  for (int off = 128; off > 0; off >>= 1) {
    if (tid < off) { s1[tid] += s1[tid + off]; s2[tid] += s2[tid + off]; }
    __syncthreads();
  }
  const float mean = s1[0] * (1.0f / Dm);
  const float var  = s2[0] * (1.0f / Dm) - mean * mean;
  const float rstd = rsqrtf(var + 1e-5f);
#pragma unroll
  for (int i = 0; i < 4; ++i) {
    int e = tid + i * 256;
    float hn = (xv[i] - mean) * rstd * g[e] + b[e];
    if (outF) outF[row * (long)Dm + e] = hn;
    outB[row * (long)Dm + e] = (__bf16)hn;
  }
}

// ---------------- column mean: first[r] = mean_n XWvT[r][n], r = b*D+e --------
__global__ __launch_bounds__(256) void colmean_kernel(
    const __bf16* __restrict__ xwvT, float* __restrict__ first) {
  __shared__ float s1[256];
  const long row = blockIdx.x;
  const int  tid = threadIdx.x;
  const __bf16* p = xwvT + row * (long)Sm;
  float s = 0.f;
#pragma unroll
  for (int i = 0; i < 2; ++i) {
    v8bf c = *(const v8bf*)(p + (tid + i * 256) * 8);
#pragma unroll
    for (int j = 0; j < 8; ++j) s += (float)c[j];
  }
  s1[tid] = s;
  __syncthreads();
  for (int off = 128; off > 0; off >>= 1) {
    if (tid < off) s1[tid] += s1[tid + off];
    __syncthreads();
  }
  if (tid == 0) first[row] = s1[0] * (1.0f / Sm);
}

// ---------------- f32 -> bf16 convert ----------------------------------------
__global__ __launch_bounds__(256) void cvt_bf16_kernel(
    const float* __restrict__ in, __bf16* __restrict__ out, long n) {
  long i = (long)blockIdx.x * blockDim.x + threadIdx.x;
  long stride = (long)gridDim.x * blockDim.x;
  for (; i < n; i += stride) out[i] = (__bf16)in[i];
}

// -----------------------------------------------------------------------------
extern "C" void kernel_launch(void* const* d_in, const int* in_sizes, int n_in,
                              void* d_out, int out_size, void* d_ws, size_t ws_size,
                              hipStream_t stream) {
  (void)in_sizes; (void)n_in; (void)out_size; (void)ws_size;
  const float* x     = (const float*)d_in[0];
  const float* W_v   = (const float*)d_in[1];
  const float* theta = (const float*)d_in[2];
  const float* ln1_g = (const float*)d_in[3];
  const float* ln1_b = (const float*)d_in[4];
  const float* ln2_g = (const float*)d_in[5];
  const float* ln2_b = (const float*)d_in[6];
  const float* w1    = (const float*)d_in[7];
  const float* b1    = (const float*)d_in[8];
  const float* w2    = (const float*)d_in[9];
  const float* b2    = (const float*)d_in[10];
  float* out = (float*)d_out;

  char* w = (char*)d_ws;
  const size_t off_hf32  = 0;                                  // 64 MB (dies after call D)
  const size_t off_XWvT  = off_hf32  + (size_t)ROWS * Dm * 4;  // 32 MB (dies after colmean/C)
  const size_t off_XthT  = off_XWvT  + (size_t)Bm_ * Dm * Sm * 2;
  const size_t off_MT    = off_XthT  + (size_t)Bm_ * Dm * Sm * 2;
  const size_t off_first = off_MT    + (size_t)Bm_ * Dm * Dm * 2;
  const size_t off_hbf   = off_first + (size_t)Bm_ * Dm * 4;
  const size_t off_Xnew  = off_hbf   + (size_t)ROWS * Dm * 2;
  const size_t off_h2    = off_Xnew  + (size_t)ROWS * Dm * 4;
  const size_t off_wv    = off_h2    + (size_t)ROWS * Dm * 2;
  const size_t off_th    = off_wv    + (size_t)Dm * Dm * 2;
  const size_t off_w1    = off_th    + (size_t)Dm * Dm * 2;
  const size_t off_w2    = off_w1    + (size_t)4 * Dm * Dm * 2;
  const size_t off_a1    = 0;  // 128 MB, aliases hf32+XWvT+XthT (all dead by then)

  float*  hf32  = (float*)(w + off_hf32);
  __bf16* XWvT  = (__bf16*)(w + off_XWvT);
  __bf16* XthT  = (__bf16*)(w + off_XthT);
  __bf16* MT    = (__bf16*)(w + off_MT);
  float*  first = (float*)(w + off_first);
  __bf16* hbf   = (__bf16*)(w + off_hbf);
  float*  Xnew  = (float*)(w + off_Xnew);
  __bf16* h2    = (__bf16*)(w + off_h2);
  __bf16* wvb   = (__bf16*)(w + off_wv);
  __bf16* thb   = (__bf16*)(w + off_th);
  __bf16* w1b   = (__bf16*)(w + off_w1);
  __bf16* w2b   = (__bf16*)(w + off_w2);
  __bf16* a1    = (__bf16*)(w + off_a1);

  // 0) weights -> bf16
  cvt_bf16_kernel<<<1024, 256, 0, stream>>>(W_v,   wvb, (long)Dm * Dm);
  cvt_bf16_kernel<<<1024, 256, 0, stream>>>(theta, thb, (long)Dm * Dm);
  cvt_bf16_kernel<<<4096, 256, 0, stream>>>(w1,    w1b, (long)4 * Dm * Dm);
  cvt_bf16_kernel<<<4096, 256, 0, stream>>>(w2,    w2b, (long)4 * Dm * Dm);

  // 1) LN1 -> h (f32 + bf16)
  ln_kernel<<<ROWS, 256, 0, stream>>>(x, ln1_g, ln1_b, hf32, hbf);

  const float scaleM = 1.0f / (4096.0f * 32.0f);  // 1/(n*sqrt(D))
  for (int b = 0; b < Bm_; ++b) {
    __bf16* Ab = hbf + (size_t)b * Dm;  // row n stride B*D
    // 2) XWvT[b][e][n] = (h W_v^T)^T
    gemm_bf16_wmma<EPI_BF16_T><<<dim3(Dm / BN, Sm / BM), 256, 0, stream>>>(
        Ab, Bm_ * Dm, wvb, Dm, Dm,
        nullptr, XWvT + (size_t)b * Dm * Sm, Sm, nullptr, nullptr, 0, nullptr, 1.0f);
    // 3) XthT[b][d][n] = (h theta^T)^T
    gemm_bf16_wmma<EPI_BF16_T><<<dim3(Dm / BN, Sm / BM), 256, 0, stream>>>(
        Ab, Bm_ * Dm, thb, Dm, Dm,
        nullptr, XthT + (size_t)b * Dm * Sm, Sm, nullptr, nullptr, 0, nullptr, 1.0f);
  }

  // 4) first[b][e] = mean_n XWv
  colmean_kernel<<<Bm_ * Dm, 256, 0, stream>>>(XWvT, first);

  for (int b = 0; b < Bm_; ++b) {
    // 5) MT[b][e][d] = scale * (XthT XWvT^T)^T    (K = n = 4096)
    gemm_bf16_wmma<EPI_BF16_T><<<dim3(Dm / BN, Dm / BM), 256, 0, stream>>>(
        XthT + (size_t)b * Dm * Sm, Sm, XWvT + (size_t)b * Dm * Sm, Sm, Sm,
        nullptr, MT + (size_t)b * Dm * Dm, Dm, nullptr, nullptr, 0, nullptr, scaleM);
  }
  for (int b = 0; b < Bm_; ++b) {
    // 6) Xnew = h + first + h_bf16 @ MT^T   (stored back in [S,B,D] layout)
    gemm_bf16_wmma<EPI_F32_RESID><<<dim3(Dm / BN, Sm / BM), 256, 0, stream>>>(
        hbf + (size_t)b * Dm, Bm_ * Dm, MT + (size_t)b * Dm * Dm, Dm, Dm,
        Xnew + (size_t)b * Dm, nullptr, Bm_ * Dm,
        nullptr, hf32 + (size_t)b * Dm, Bm_ * Dm, first + (size_t)b * Dm, 1.0f);
  }

  // 7) LN2 -> h2 (bf16 only)
  ln_kernel<<<ROWS, 256, 0, stream>>>(Xnew, ln2_g, ln2_b, nullptr, h2);

  // 8) MLP GEMM1: a1 = gelu(h2 @ w1^T + b1)   [16384 x 4096]
  gemm_bf16_wmma<EPI_GELU_BF16><<<dim3(4 * Dm / BN, ROWS / BM), 256, 0, stream>>>(
      h2, Dm, w1b, Dm, Dm,
      nullptr, a1, 4 * Dm, b1, nullptr, 0, nullptr, 1.0f);

  // 9) MLP GEMM2: out = Xnew + a1 @ w2^T + b2   [16384 x 1024]
  gemm_bf16_wmma<EPI_F32_RESID><<<dim3(Dm / BN, ROWS / BM), 256, 0, stream>>>(
      a1, 4 * Dm, w2b, 4 * Dm, 4 * Dm,
      out, nullptr, Dm, b2, Xnew, Dm, nullptr, 1.0f);
}
